// TwoLayerGCN_20358144983128
// MI455X (gfx1250) — matrix operand order repaired and checked by reading, not compiled, not run on hardware
//
#include <hip/hip_runtime.h>
#include <hip/hip_bf16.h>

// ---------- types for WMMA fragments ----------
typedef __attribute__((ext_vector_type(16))) __bf16          v16bf;
typedef __attribute__((ext_vector_type(8)))  float           v8f;
typedef __attribute__((ext_vector_type(8)))  unsigned short  ushort8;

union FragBF16 {
    v16bf   bf;
    ushort8 us[2];
};

__device__ __forceinline__ unsigned short f32_to_bf16_rne(float f) {
    union { float f; unsigned int u; } v; v.f = f;
    unsigned int r = v.u + 0x7FFFu + ((v.u >> 16) & 1u);   // round-to-nearest-even
    return (unsigned short)(r >> 16);
}

// ---------- small utility kernels ----------
__global__ void zero_f32_kernel(float* __restrict__ p, size_t n4) {
    size_t i = (size_t)blockIdx.x * blockDim.x + threadIdx.x;
    if (i < n4) {
        float4 z = {0.f, 0.f, 0.f, 0.f};
        *reinterpret_cast<float4*>(p + i * 4) = z;
    }
}

// float [n] -> bf16 [n], 4 elems/thread
__global__ void cvt_f32_bf16_kernel(const float* __restrict__ x,
                                    unsigned short* __restrict__ xb, size_t n4) {
    size_t i = (size_t)blockIdx.x * blockDim.x + threadIdx.x;
    if (i < n4) {
        float4 v = *reinterpret_cast<const float4*>(x + i * 4);
        unsigned short o0 = f32_to_bf16_rne(v.x);
        unsigned short o1 = f32_to_bf16_rne(v.y);
        unsigned short o2 = f32_to_bf16_rne(v.z);
        unsigned short o3 = f32_to_bf16_rne(v.w);
        unsigned long long packed =
            (unsigned long long)o0 | ((unsigned long long)o1 << 16) |
            ((unsigned long long)o2 << 32) | ((unsigned long long)o3 << 48);
        *reinterpret_cast<unsigned long long*>(xb + i * 4) = packed;
    }
}

// relu(float [n]) -> bf16 [n], 4 elems/thread
__global__ void relu_cvt_f32_bf16_kernel(const float* __restrict__ x,
                                         unsigned short* __restrict__ xb, size_t n4) {
    size_t i = (size_t)blockIdx.x * blockDim.x + threadIdx.x;
    if (i < n4) {
        float4 v = *reinterpret_cast<const float4*>(x + i * 4);
        v.x = fmaxf(v.x, 0.f); v.y = fmaxf(v.y, 0.f);
        v.z = fmaxf(v.z, 0.f); v.w = fmaxf(v.w, 0.f);
        unsigned short o0 = f32_to_bf16_rne(v.x);
        unsigned short o1 = f32_to_bf16_rne(v.y);
        unsigned short o2 = f32_to_bf16_rne(v.z);
        unsigned short o3 = f32_to_bf16_rne(v.w);
        unsigned long long packed =
            (unsigned long long)o0 | ((unsigned long long)o1 << 16) |
            ((unsigned long long)o2 << 32) | ((unsigned long long)o3 << 48);
        *reinterpret_cast<unsigned long long*>(xb + i * 4) = packed;
    }
}

// W[rows,cols] f32 row-major -> WT[cols,rows] bf16 (column-major B for contiguous K reads)
__global__ void transpose_cvt_bf16_kernel(const float* __restrict__ W,
                                          unsigned short* __restrict__ WT,
                                          int rows, int cols) {
    int idx = blockIdx.x * blockDim.x + threadIdx.x;
    if (idx < rows * cols) {
        int c = idx / rows;
        int r = idx - c * rows;
        WT[(size_t)c * rows + r] = f32_to_bf16_rne(W[(size_t)r * cols + c]);
    }
}

// ---------- WMMA bf16 GEMM: C[M,NCOLS] = A[M,K] * B[K,NCOLS]  (BT is [NCOLS,K]) ----------
// One wave computes one 16x16 output tile. Block = (NCOLS/16) waves covering all columns
// of one 16-row stripe. M must be a multiple of 16 (100000 = 16*6250).
template <int K, int NCOLS>
__global__ void gemm_bf16_wmma_kernel(const unsigned short* __restrict__ A,   // bf16 [M,K]
                                      const unsigned short* __restrict__ BT,  // bf16 [NCOLS,K]
                                      float* __restrict__ C) {
    const int lane  = threadIdx.x & 31;
    const int wave  = threadIdx.x >> 5;       // = N tile index
    const int m15   = lane & 15;
    const int h     = lane >> 4;              // lane-half selector per ISA layout
    const size_t rowA = (size_t)blockIdx.x * 16 + m15;          // A row for this lane
    const size_t colB = (size_t)wave * 16 + m15;                // B column for this lane

    const unsigned short* aRow = A  + rowA * K;
    const unsigned short* bCol = BT + colB * K;

    v8f acc = {};
#pragma unroll
    for (int k0 = 0; k0 < K; k0 += 32) {
        FragBF16 a, b;
        // A 16x32 bf16 fragment: VGPR0-3 = K[h*8 .. h*8+7], VGPR4-7 = K[16+h*8 .. 16+h*8+7]
        a.us[0] = *reinterpret_cast<const ushort8*>(aRow + k0 + h * 8);
        a.us[1] = *reinterpret_cast<const ushort8*>(aRow + k0 + 16 + h * 8);
        // B 32x16 bf16 fragment: VGPR0-3 = K[h*8 .. h*8+7], VGPR4-7 = K[16+h*8 .. 16+h*8+7] of its column
        b.us[0] = *reinterpret_cast<const ushort8*>(bCol + k0 + h * 8);
        b.us[1] = *reinterpret_cast<const ushort8*>(bCol + k0 + 16 + h * 8);
        acc = __builtin_amdgcn_wmma_f32_16x16x32_bf16(
            /*neg_a=*/false, a.bf, /*neg_b=*/false, b.bf,
            /*c_mod=*/(short)0, acc, /*reuse_a=*/false, /*reuse_b=*/false);
    }

    // C/D layout: lane n = N, VGPR r -> M = r + 8*h
    const size_t nbase = (size_t)wave * 16 + m15;
    const size_t mbase = (size_t)blockIdx.x * 16 + h * 8;
#pragma unroll
    for (int r = 0; r < 8; ++r)
        C[(mbase + r) * NCOLS + nbase] = acc[r];
}

// ---------- SpMM scatter: out[dst] += H[src], F features, float4 per thread ----------
template <int F>
__global__ void spmm_scatter_kernel(const float* __restrict__ H,
                                    const long long* __restrict__ src,
                                    const long long* __restrict__ dst,
                                    float* __restrict__ out, long long nEdges) {
    const long long idx = (long long)blockIdx.x * blockDim.x + threadIdx.x;
    const long long e   = idx / (F / 4);
    const int       c   = (int)(idx % (F / 4)) * 4;
    if (e < nEdges) {
        const long long s = src[e];
        const long long d = dst[e];
        float4 v = *reinterpret_cast<const float4*>(H + s * F + c);
        float* o = out + d * F + c;
        atomicAdd(o + 0, v.x);
        atomicAdd(o + 1, v.y);
        atomicAdd(o + 2, v.z);
        atomicAdd(o + 3, v.w);
    }
}

// ---------- launcher ----------
extern "C" void kernel_launch(void* const* d_in, const int* in_sizes, int n_in,
                              void* d_out, int out_size, void* d_ws, size_t ws_size,
                              hipStream_t stream) {
    const float*     x   = (const float*)d_in[0];
    const long long* esr = (const long long*)d_in[1];
    const long long* eds = (const long long*)d_in[2];
    const float*     W1  = (const float*)d_in[3];
    const float*     W2  = (const float*)d_in[4];
    float*           out = (float*)d_out;

    const size_t N = (size_t)in_sizes[0] / 256;   // 100000 nodes (multiple of 16)
    const long long E = (long long)in_sizes[1];   // 3.2M edges

    // workspace layout (regions aliased across phases)
    char* ws = (char*)d_ws;
    unsigned short* W1T = (unsigned short*)(ws);           // [128,256] bf16  (64 KB)
    unsigned short* W2T = (unsigned short*)(ws + 65536);   // [64,128]  bf16  (16 KB)
    char* r1 = ws + 131072;                                // 51.2 MB region
    char* r2 = r1 + N * 512;                               // 51.2 MB region
    unsigned short* xb   = (unsigned short*)r1;            // x bf16   [N,256]
    float*          h1   = (float*)r2;                     // x@W1     [N,128]
    float*          agg1 = (float*)r1;                     // spmm1    [N,128] (xb dead)
    unsigned short* hrb  = (unsigned short*)r2;            // relu bf16[N,128] (h1 dead)
    float*          h2   = (float*)r1;                     // @W2      [N,64]  (agg1 dead)

    const int T = 256;

    // 1) weights: transpose + convert to bf16
    transpose_cvt_bf16_kernel<<<(256 * 128 + T - 1) / T, T, 0, stream>>>(W1, W1T, 256, 128);
    transpose_cvt_bf16_kernel<<<(128 * 64 + T - 1) / T, T, 0, stream>>>(W2, W2T, 128, 64);

    // 2) x -> bf16
    {
        size_t n4 = N * 256 / 4;
        cvt_f32_bf16_kernel<<<(unsigned)((n4 + T - 1) / T), T, 0, stream>>>(x, xb, n4);
    }

    // 3) h1 = x @ W1   (WMMA bf16, f32 accumulate)
    gemm_bf16_wmma_kernel<256, 128><<<(unsigned)(N / 16), 8 * 32, 0, stream>>>(xb, W1T, h1);

    // 4) agg1 = segment_sum(h1[src] -> dst)
    {
        size_t n4 = N * 128 / 4;
        zero_f32_kernel<<<(unsigned)((n4 + T - 1) / T), T, 0, stream>>>(agg1, n4);
        long long work = E * (128 / 4);
        spmm_scatter_kernel<128><<<(unsigned)((work + T - 1) / T), T, 0, stream>>>(h1, esr, eds, agg1, E);
    }

    // 5) relu + bf16 convert
    {
        size_t n4 = N * 128 / 4;
        relu_cvt_f32_bf16_kernel<<<(unsigned)((n4 + T - 1) / T), T, 0, stream>>>(agg1, hrb, n4);
    }

    // 6) h2 = relu(agg1) @ W2
    gemm_bf16_wmma_kernel<128, 64><<<(unsigned)(N / 16), 4 * 32, 0, stream>>>(hrb, W2T, h2);

    // 7) out = segment_sum(h2[src] -> dst)
    {
        size_t n4 = N * 64 / 4;
        zero_f32_kernel<<<(unsigned)((n4 + T - 1) / T), T, 0, stream>>>(out, n4);
        long long work = E * (64 / 4);
        spmm_scatter_kernel<64><<<(unsigned)((work + T - 1) / T), T, 0, stream>>>(h2, esr, eds, out, E);
    }
}
